// TrainRouting_68324339745381
// MI455X (gfx1250) — matrix operand order repaired and checked by reading, not compiled, not run on hardware
//
#include <hip/hip_runtime.h>

#define B_  64
#define P_  2304
#define D1_ 16
#define S_  32
#define D2_ 32

typedef _Float16 h8   __attribute__((ext_vector_type(8)));
typedef _Float16 v16h __attribute__((ext_vector_type(16)));
typedef float    v8f  __attribute__((ext_vector_type(8)));

static __device__ __forceinline__ v16h cat16(h8 a, h8 b) {
    return __builtin_shufflevector(a, b, 0,1,2,3,4,5,6,7,8,9,10,11,12,13,14,15);
}
static __device__ __forceinline__ v8f wmma_f16(v16h a, v16h b, v8f c) {
    // D = A(16x32 f16) * B(32x16 f16) + C(16x16 f32)
    return __builtin_amdgcn_wmma_f32_16x16x32_f16(false, a, false, b, (short)0, c, false, false);
}

// ---------------------------------------------------------------------------
// Kernel 1: u = squash(caps1_raw) over D1; emit f16 in two layouts:
//   uA[p][b][k]  (A operand for ws GEMM)
//   uT[p][k][b]  (A operand for agree GEMM, K = batch)
// Also zero routing logits b_route[P,S].
// Grid: 576 x 256 == B*P threads exactly.
// ---------------------------------------------------------------------------
__global__ void k_prep(const float* __restrict__ caps1,
                       _Float16* __restrict__ uA,
                       _Float16* __restrict__ uT,
                       float* __restrict__ b_route) {
    int idx = blockIdx.x * blockDim.x + threadIdx.x;       // b*P + p
    int b = idx / P_;
    int p = idx - b * P_;

    const float4* src = (const float4*)(caps1 + (size_t)idx * 16);
    float4 x0 = src[0], x1 = src[1], x2 = src[2], x3 = src[3];
    float f[16] = { x0.x,x0.y,x0.z,x0.w, x1.x,x1.y,x1.z,x1.w,
                    x2.x,x2.y,x2.z,x2.w, x3.x,x3.y,x3.z,x3.w };
    float sq = 0.f;
    #pragma unroll
    for (int k = 0; k < 16; ++k) sq += f[k] * f[k];
    float sc = (sq / (1.0f + sq)) * rsqrtf(sq + 1e-7f);

    h8 lo, hi;
    #pragma unroll
    for (int k = 0; k < 8; ++k) { lo[k] = (_Float16)(f[k] * sc); hi[k] = (_Float16)(f[8 + k] * sc); }

    _Float16* dA = uA + ((size_t)((p << 6) + b) << 4);     // (p*64+b)*16
    *(h8*)dA       = lo;
    *(h8*)(dA + 8) = hi;

    #pragma unroll
    for (int k = 0; k < 8; ++k) {
        uT[(((p << 4) + k)     << 6) + b] = lo[k];         // (p*16+k)*64 + b
        uT[(((p << 4) + 8 + k) << 6) + b] = hi[k];
    }

    if (idx < P_ * S_) b_route[idx] = 0.0f;
}

// ---------------------------------------------------------------------------
// Kernel 2: rw[p,s] = softmax_s(b_route[p,:]), one wave per p (lane = s).
// Also zeroes the ws accumulator (written by atomics in k_ws).
// Grid: 288 x 256 == P*S threads exactly.
// ---------------------------------------------------------------------------
__global__ void k_rw(const float* __restrict__ b_route,
                     float* __restrict__ rw,
                     float* __restrict__ wsum) {
    int tid = blockIdx.x * 256 + threadIdx.x;              // p*32 + s
    if (tid < B_ * S_ * D2_) wsum[tid] = 0.0f;

    float x = b_route[tid];
    float m = x;
    #pragma unroll
    for (int d = 16; d >= 1; d >>= 1) m = fmaxf(m, __shfl_xor(m, d, 32));
    float e = __expf(x - m);
    float s = e;
    #pragma unroll
    for (int d = 16; d >= 1; d >>= 1) s += __shfl_xor(s, d, 32);
    rw[tid] = e / s;
}

// ---------------------------------------------------------------------------
// Kernel 3: ws[b,s,d] += sum_{p,k} (rw[p,s]*W[p,s,d,k]) * u[b,p,k]
// WMMA: M=b-tile(16), N=d-tile(16), K=32 = two p's x 16 k.
// Wave = (p-slice of 64 p's, s, b-tile, d-tile); 32 K-chunks per wave.
// Grid: 1152 x 256 == 9216 waves exactly (EXEC all ones for WMMA).
// ---------------------------------------------------------------------------
__global__ void __launch_bounds__(256) k_ws(const _Float16* __restrict__ uA,
                                            const float* __restrict__ W,
                                            const float* __restrict__ rw,
                                            float* __restrict__ wsum) {
    int lane = threadIdx.x & 31;
    int w    = blockIdx.x * 8 + (threadIdx.x >> 5);
    int slice = w >> 8;            // 0..35  (64 p's each)
    int rem   = w & 255;
    int s  = rem >> 3;             // 0..31
    int bt = (rem >> 1) & 3;       // 0..3
    int dt = rem & 1;              // 0..1

    bool lo = lane < 16;
    int  m  = lane & 15;
    int  b_row = bt * 16 + m;      // A row (M)
    int  d_col = dt * 16 + m;      // B column (N)
    int  koff  = lo ? 0 : 8;       // A layout: low lanes K0-15 (p0), high lanes K16-31 (p1)

    v8f acc = {};
    for (int ch = 0; ch < 32; ++ch) {
        int p0 = slice * 64 + ch * 2;
        int p1 = p0 + 1;

        // A operand: low lanes u[b_row, p0, 0..15], high lanes u[b_row, p1, 0..15]
        const _Float16* a0 = uA + (((size_t)(((lo ? p0 : p1) << 6) + b_row)) << 4);
        const _Float16* a1 = a0 + 8;
        v16h A = cat16(*(const h8*)a0, *(const h8*)a1);

        // B operand: low lanes hold K0-15 (p0), high lanes K16-31 (p1);
        // 16 contiguous f32 of W[p,s,d_col,0:16], scaled by rw[p,s], cvt->f16.
        int pb = lo ? p0 : p1;
        float r = rw[(pb << 5) + s];
        const float4* wp = (const float4*)(W + (size_t)((pb << 5) + s) * 512 + (d_col << 4));
        float4 w0 = wp[0], w1 = wp[1], w2 = wp[2], w3 = wp[3];
        v16h Bv;
        Bv[0]=(_Float16)(w0.x*r); Bv[1]=(_Float16)(w0.y*r); Bv[2]=(_Float16)(w0.z*r); Bv[3]=(_Float16)(w0.w*r);
        Bv[4]=(_Float16)(w1.x*r); Bv[5]=(_Float16)(w1.y*r); Bv[6]=(_Float16)(w1.z*r); Bv[7]=(_Float16)(w1.w*r);
        Bv[8]=(_Float16)(w2.x*r); Bv[9]=(_Float16)(w2.y*r); Bv[10]=(_Float16)(w2.z*r); Bv[11]=(_Float16)(w2.w*r);
        Bv[12]=(_Float16)(w3.x*r); Bv[13]=(_Float16)(w3.y*r); Bv[14]=(_Float16)(w3.z*r); Bv[15]=(_Float16)(w3.w*r);

        acc = wmma_f16(A, Bv, acc);
    }

    // C layout: lanes 0-15 -> rows M=0..7, lanes 16-31 -> rows M=8..15; N = lane&15.
    int bb = bt * 16 + (lo ? 0 : 8);
    float* dst = wsum + (((bb << 5) + s) << 5) + d_col;    // (b*32+s)*32 + d
    #pragma unroll
    for (int rr = 0; rr < 8; ++rr)
        unsafeAtomicAdd(dst + (rr << 10), acc[rr]);        // +1024 per b-row
}

// ---------------------------------------------------------------------------
// Kernel 4: squash over D2 per (b,s) row; wave per row (lane = d).
// Final iteration writes f32 output; intermediate writes v_T f16 [S,D2,B].
// Grid: 256 x 256 == B*S*D2 threads exactly.
// ---------------------------------------------------------------------------
__global__ void k_squash(const float* __restrict__ wsum,
                         float* __restrict__ outF,
                         _Float16* __restrict__ vT) {
    int tid = blockIdx.x * 256 + threadIdx.x;              // (b*32+s)*32 + d
    int d   = tid & 31;
    int row = tid >> 5;                                    // b*32 + s
    float x = wsum[tid];
    float sq = x * x;
    #pragma unroll
    for (int k = 16; k >= 1; k >>= 1) sq += __shfl_xor(sq, k, 32);
    float val = x * (sq / (1.0f + sq)) * rsqrtf(sq + 1e-7f);
    if (outF) outF[tid] = val;
    if (vT) {
        int b = row >> 5, s = row & 31;
        vT[((((s << 5) + d)) << 6) + b] = (_Float16)val;   // (s*32+d)*64 + b
    }
}

// ---------------------------------------------------------------------------
// Kernel 5: b_route[p,s] += (1/B) * sum_{d,k} W[p,s,d,k] * C_p[k,d]
// with C_p = u_T[p]^T(16xB) x v_T[s](Bx32): WMMA M=k(16), N=d(2 tiles), K=b(2x32).
// Wave per (p,s). Grid: 9216 x 256 == 73728 waves exactly.
// ---------------------------------------------------------------------------
__global__ void __launch_bounds__(256) k_agree(const _Float16* __restrict__ uT,
                                               const _Float16* __restrict__ vT,
                                               const float* __restrict__ W,
                                               float* __restrict__ b_route) {
    int lane = threadIdx.x & 31;
    int w = blockIdx.x * 8 + (threadIdx.x >> 5);
    int p = w >> 5;
    int s = w & 31;
    bool lo = lane < 16;
    int  m  = lane & 15;

    v8f c0 = {}, c1 = {};
    #pragma unroll
    for (int ch = 0; ch < 2; ++ch) {
        int bb = ch << 5;
        // A: row = k-index m, K = batch within chunk
        const _Float16* au = uT + ((((p << 4) + m)) << 6) + bb;
        v16h A = cat16(*(const h8*)(au + (lo ? 0 : 16)), *(const h8*)(au + (lo ? 8 : 24)));
        // B d-tile 0: column n=m -> d=m, K = batch
        const _Float16* v0 = vT + ((((s << 5) + m)) << 6) + bb + (lo ? 0 : 16);
        c0 = wmma_f16(A, cat16(*(const h8*)v0, *(const h8*)(v0 + 8)), c0);
        // B d-tile 1: d = 16+m
        const _Float16* v1 = vT + ((((s << 5) + 16 + m)) << 6) + bb + (lo ? 0 : 16);
        c1 = wmma_f16(A, cat16(*(const h8*)v1, *(const h8*)(v1 + 8)), c1);
    }

    // Elementwise dot with W: lane holds C[k = koff+r, d = (tile)*16 + m]
    const float* Wps = W + (size_t)((p << 5) + s) * 512;   // W[p][s][d][k]
    int koff = lo ? 0 : 8;
    const float4* q0 = (const float4*)(Wps + (m << 4) + koff);
    const float4* q1 = (const float4*)(Wps + ((16 + m) << 4) + koff);
    float4 a0 = q0[0], a1 = q0[1], b0 = q1[0], b1 = q1[1];
    float part = c0[0]*a0.x + c0[1]*a0.y + c0[2]*a0.z + c0[3]*a0.w
               + c0[4]*a1.x + c0[5]*a1.y + c0[6]*a1.z + c0[7]*a1.w
               + c1[0]*b0.x + c1[1]*b0.y + c1[2]*b0.z + c1[3]*b0.w
               + c1[4]*b1.x + c1[5]*b1.y + c1[6]*b1.z + c1[7]*b1.w;
    #pragma unroll
    for (int k = 16; k >= 1; k >>= 1) part += __shfl_xor(part, k, 32);
    if (lane == 0) b_route[(p << 5) + s] += part * (1.0f / 64.0f);
}

// ---------------------------------------------------------------------------
extern "C" void kernel_launch(void* const* d_in, const int* in_sizes, int n_in,
                              void* d_out, int out_size, void* d_ws, size_t ws_size,
                              hipStream_t stream) {
    const float* caps1 = (const float*)d_in[0];            // [B,P,16]
    const float* W     = (const float*)d_in[1];            // [P,S,D2,D1]
    float* out = (float*)d_out;                            // [B,1,S,D2,1] flat

    char* wsb = (char*)d_ws;
    size_t off = 0;
    _Float16* uA = (_Float16*)(wsb + off); off += (size_t)P_ * B_ * 16 * 2;
    _Float16* uT = (_Float16*)(wsb + off); off += (size_t)P_ * B_ * 16 * 2;
    _Float16* vT = (_Float16*)(wsb + off); off += (size_t)S_ * D2_ * B_ * 2;
    float* rw      = (float*)(wsb + off);  off += (size_t)P_ * S_ * 4;
    float* b_route = (float*)(wsb + off);  off += (size_t)P_ * S_ * 4;
    float* wsum    = (float*)(wsb + off);  off += (size_t)B_ * S_ * D2_ * 4;
    (void)ws_size; (void)in_sizes; (void)n_in; (void)out_size;

    k_prep<<<576, 256, 0, stream>>>(caps1, uA, uT, b_route);
    for (int it = 0; it < 3; ++it) {
        k_rw<<<288, 256, 0, stream>>>(b_route, rw, wsum);
        k_ws<<<1152, 256, 0, stream>>>(uA, W, rw, wsum);
        if (it < 2) {
            k_squash<<<256, 256, 0, stream>>>(wsum, nullptr, vT);
            k_agree<<<9216, 256, 0, stream>>>(uT, vT, W, b_route);
        } else {
            k_squash<<<256, 256, 0, stream>>>(wsum, out, nullptr);
        }
    }
}
